// NMREmbedPatchAttention_29076928594300
// MI455X (gfx1250) — matrix operand order repaired and checked by reading, not compiled, not run on hardware
//
#include <hip/hip_runtime.h>

typedef __bf16 bf16_t;
typedef __attribute__((ext_vector_type(16))) __bf16 v16bf;
typedef __attribute__((ext_vector_type(8)))  __bf16 v8bf;
typedef __attribute__((ext_vector_type(8)))  float   v8f;
typedef unsigned int u32;
typedef __attribute__((ext_vector_type(4))) unsigned int v4u;
typedef __attribute__((ext_vector_type(8))) int          v8i;
typedef __attribute__((ext_vector_type(4))) int          v4i;

#define D_MODEL 256
#define PATCH   8
#define DK      32
#define LDA     272   // padded LDS row stride (bf16) for kernel A buffers
#define KC      256   // K-chunk staged per TDM transfer in kernel B
#define LDB_B   264   // staged-A row stride (bf16): 256 data + 8 pad (4 DWORDs)

#if __has_builtin(__builtin_amdgcn_tensor_load_to_lds) && __has_builtin(__builtin_amdgcn_s_wait_tensorcnt)
#define USE_TDM 1
// Toolchain arity split (probe-verified): ROCm 7.2 / clang-22 = 5 args,
// amdgpu-toolchain / clang-23 = 6 args (extra int32x8 group before cpol).
#if defined(__clang_major__) && __clang_major__ >= 23
#define TDM_LOAD(g0, g1, g2, g3)                                        \
  do {                                                                  \
    v8i _z8 = {0, 0, 0, 0, 0, 0, 0, 0};                                 \
    __builtin_amdgcn_tensor_load_to_lds((g0), (g1), (g2), (g3), _z8, 0);\
  } while (0)
#else
#define TDM_LOAD(g0, g1, g2, g3)                                        \
  __builtin_amdgcn_tensor_load_to_lds((g0), (g1), (g2), (g3), 0)
#endif
#else
#define USE_TDM 0
#endif

// ---- WMMA fragment loaders (CDNA5 16-bit layouts, CDNA5 ISA §7.12.2) ----
// A (16xK row-major): lane&15 -> M; lanes 0-15 hold K {0..7,16..23}+kb, lanes 16-31 {8..15,24..31}+kb.
__device__ __forceinline__ v16bf load_a_frag(const bf16_t* A, int lda, int lane, int kb) {
  const int row  = lane & 15;
  const int koff = (lane >> 4) << 3;            // 0 or 8
  const bf16_t* p = A + row * lda + kb + koff;
  v8bf lo = *(const v8bf*)(p);                  // K = kb+koff .. +7
  v8bf hi = *(const v8bf*)(p + 16);             // K = kb+16+koff .. +7
  return __builtin_shufflevector(lo, hi, 0,1,2,3,4,5,6,7,8,9,10,11,12,13,14,15);
}
// B (32x16 tile of row-major weight [K][N]): lane&15 -> N; lanes 0-15 hold K {0..7,16..23}+kb, lanes 16-31 {8..15,24..31}+kb.
__device__ __forceinline__ v16bf load_b_frag(const bf16_t* B, int ldb, int lane, int kb, int nb) {
  const int n    = nb + (lane & 15);
  const int koff = (lane >> 4) << 3;
  v16bf r;
  #pragma unroll
  for (int s = 0; s < 8; ++s) {
    r[s]     = B[(kb + koff + s) * ldb + n];
    r[s + 8] = B[(kb + 16 + koff + s) * ldb + n];
  }
  return r;
}

// 16x32 output strip: A from LDS, B = bf16 weight [256][256], +bias, bf16 result to LDS.
__device__ __forceinline__ void gemm16_lds(const bf16_t* Alds, const bf16_t* Bw,
                                           const float* __restrict__ bias, bf16_t* Olds,
                                           int lane, int nbase) {
  v8f acc0 = {}; v8f acc1 = {};
  #pragma unroll
  for (int kb = 0; kb < D_MODEL; kb += 32) {
    v16bf a  = load_a_frag(Alds, LDA, lane, kb);
    v16bf b0 = load_b_frag(Bw, D_MODEL, lane, kb, nbase);
    v16bf b1 = load_b_frag(Bw, D_MODEL, lane, kb, nbase + 16);
    acc0 = __builtin_amdgcn_wmma_f32_16x16x32_bf16(false, a, false, b0, (short)0, acc0, false, false);
    acc1 = __builtin_amdgcn_wmma_f32_16x16x32_bf16(false, a, false, b1, (short)0, acc1, false, false);
  }
  const int col = nbase + (lane & 15);
  const int rb  = (lane >> 4) << 3;
  const float bi0 = bias[col];
  const float bi1 = bias[col + 16];
  #pragma unroll
  for (int r = 0; r < 8; ++r) {
    Olds[(rb + r) * LDA + col]      = (bf16_t)(acc0[r] + bi0);
    Olds[(rb + r) * LDA + col + 16] = (bf16_t)(acc1[r] + bi1);
  }
}

// ---- Kernel 0: fp32 -> bf16 weight conversion ----
__global__ void cvt_f32_bf16_kernel(const float* __restrict__ src, bf16_t* __restrict__ dst, int n) {
  int i = blockIdx.x * blockDim.x + threadIdx.x;
  if (i < n) dst[i] = (bf16_t)src[i];
}

// ---- Kernel A: fused embed-MLP + QKV + windowed attention + Wo, per 16-token tile ----
__global__ __launch_bounds__(256)
void fused_embed_attn_kernel(const float* __restrict__ spec,
                             const float* __restrict__ W1, const float* __restrict__ b1,
                             const float* __restrict__ b2,
                             const float* __restrict__ bq, const float* __restrict__ bk,
                             const float* __restrict__ bv, const float* __restrict__ bo,
                             const bf16_t* __restrict__ W2b,
                             const bf16_t* __restrict__ Wqb, const bf16_t* __restrict__ Wkb,
                             const bf16_t* __restrict__ Wvb, const bf16_t* __restrict__ Wob,
                             bf16_t* __restrict__ Oout) {
  __shared__ __align__(32) bf16_t sH [16 * LDA];
  __shared__ __align__(32) bf16_t sE [16 * LDA];
  __shared__ __align__(32) bf16_t sQ [16 * LDA];
  __shared__ __align__(32) bf16_t sK [16 * LDA];
  __shared__ __align__(32) bf16_t sV [16 * LDA];
  __shared__ __align__(32) bf16_t sOh[16 * LDA];

  const int tid  = threadIdx.x;
  const int lane = tid & 31;
  const int wave = tid >> 5;
  const int t0   = blockIdx.x * 16;     // 16 tokens = 2 complete windows

  // Stage 1: H = relu(spec[t] * W1 + b1), 16x256 -> LDS bf16
  {
    const int row = tid >> 4;
    const int c0  = (tid & 15) * 16;
    const float s = spec[t0 + row];
    #pragma unroll
    for (int j = 0; j < 16; ++j) {
      float h = fmaf(s, W1[c0 + j], b1[c0 + j]);
      sH[row * LDA + c0 + j] = (bf16_t)fmaxf(h, 0.0f);
    }
  }
  __syncthreads();

  const int nbase = wave * 32;          // each of 8 waves owns a 32-col strip

  gemm16_lds(sH, W2b, b2, sE, lane, nbase);          // E = H @ W2 + b2
  __syncthreads();
  gemm16_lds(sE, Wqb, bq, sQ, lane, nbase);          // Q/K/V
  gemm16_lds(sE, Wkb, bk, sK, lane, nbase);
  gemm16_lds(sE, Wvb, bv, sV, lane, nbase);
  __syncthreads();

  // Windowed attention: 2 windows x 8 heads x 8 query rows = 128 threads (~1% of FLOPs)
  if (tid < 128) {
    const int w = tid >> 6;
    const int h = (tid >> 3) & 7;
    const int q = tid & 7;
    const bf16_t* qp = &sQ[(w * PATCH + q) * LDA + h * DK];
    float sc[PATCH];
    float m = -1e30f;
    #pragma unroll
    for (int kk = 0; kk < PATCH; ++kk) {
      const bf16_t* kp = &sK[(w * PATCH + kk) * LDA + h * DK];
      float d = 0.0f;
      #pragma unroll
      for (int e = 0; e < DK; ++e) d = fmaf((float)qp[e], (float)kp[e], d);
      sc[kk] = d * 0.17677669529663687f;   // 1/sqrt(32)
      m = fmaxf(m, sc[kk]);
    }
    float sum = 0.0f;
    #pragma unroll
    for (int kk = 0; kk < PATCH; ++kk) { sc[kk] = __expf(sc[kk] - m); sum += sc[kk]; }
    const float inv = 1.0f / sum;
    bf16_t* op = &sOh[(w * PATCH + q) * LDA + h * DK];
    #pragma unroll
    for (int e = 0; e < DK; ++e) {
      float o = 0.0f;
      #pragma unroll
      for (int kk = 0; kk < PATCH; ++kk)
        o = fmaf(sc[kk], (float)sV[(w * PATCH + kk) * LDA + h * DK + e], o);
      op[e] = (bf16_t)(o * inv);
    }
  }
  __syncthreads();

  // O = Oh @ Wo + bo -> global bf16 [T][256] (== [NW][2048] for kernel B)
  {
    v8f acc0 = {}; v8f acc1 = {};
    #pragma unroll
    for (int kb = 0; kb < D_MODEL; kb += 32) {
      v16bf a  = load_a_frag(sOh, LDA, lane, kb);
      v16bf b0 = load_b_frag(Wob, D_MODEL, lane, kb, nbase);
      v16bf b1 = load_b_frag(Wob, D_MODEL, lane, kb, nbase + 16);
      acc0 = __builtin_amdgcn_wmma_f32_16x16x32_bf16(false, a, false, b0, (short)0, acc0, false, false);
      acc1 = __builtin_amdgcn_wmma_f32_16x16x32_bf16(false, a, false, b1, (short)0, acc1, false, false);
    }
    const int col = nbase + (lane & 15);
    const int rb  = (lane >> 4) << 3;
    const float bi0 = bo[col];
    const float bi1 = bo[col + 16];
    #pragma unroll
    for (int r = 0; r < 8; ++r) {
      Oout[(size_t)(t0 + rb + r) * D_MODEL + col]      = (bf16_t)(acc0[r] + bi0);
      Oout[(size_t)(t0 + rb + r) * D_MODEL + col + 16] = (bf16_t)(acc1[r] + bi1);
    }
  }
}

// ---- Kernel B: out = O[NW,2048] @ Wp[2048,256] + bp (fp32 out) ----
// A tile (16x2048 bf16, the only HBM-streaming operand: ~105 MB) is staged into LDS in
// 16xKC chunks by the Tensor Data Mover (double-buffered), instead of 8 waves each
// re-fetching identical rows from global. Wp stays in L2 (192 MB) and is read directly.
__global__ __launch_bounds__(256)
void patch_proj_kernel(const bf16_t* __restrict__ O2d, const bf16_t* __restrict__ Wpb,
                       const float* __restrict__ bp, float* __restrict__ out) {
  __shared__ __align__(64) bf16_t sA[2][16 * LDB_B];
  const int tid   = threadIdx.x;
  const int lane  = tid & 31;
  const int wave  = tid >> 5;
  const int m0    = blockIdx.x * 16;
  const int nbase = wave * 32;
  const bf16_t* Abase = O2d + (size_t)m0 * 2048;

#if USE_TDM
  // One TDM descriptor per chunk: 2D tile 16 rows x KC cols of 2-byte elements,
  // tensor_dim0_stride = 2048, LDS padding 4 DWORDs every 128 DWORDs -> row stride LDB_B.
  auto issue = [&](int c) {
    if (wave == 0) {
      const unsigned lds_off = (unsigned)(size_t)(&sA[c & 1][0]);   // AS3 offset = low 32 bits
      const unsigned long long ga = (unsigned long long)(size_t)(Abase + c * KC);
      v4u g0;
      g0[0] = 1u;                                            // count=1, user descriptor
      g0[1] = lds_off;                                       // lds_addr (bytes)
      g0[2] = (u32)(ga & 0xffffffffu);                       // global_addr[31:0]
      g0[3] = (u32)((ga >> 32) & 0x1ffffffu) | (2u << 30);   // global_addr[56:32] | type=2
      v8i g1;
      g1[0] = (1 << 16)        // data_size = 1 -> 2-byte elements
            | (1 << 20)        // pad_enable
            | (6 << 22)        // pad_interval code 6 = every 128 DWORDs (one 512B row)
            | (3 << 25);       // pad_amount  code 3 = 4 DWORDs (8 bf16) -> stride 264
      g1[1] = (2048 & 0xffff) << 16;                 // tensor_dim0[15:0]   (bits 63:48)
      g1[2] = (2048 >> 16) | (16 << 16);             // tensor_dim0[31:16], tensor_dim1[15:0]=16
      g1[3] = (KC << 16);                            // tensor_dim1[31:16]=0, tile_dim0=KC
      g1[4] = 16;                                    // tile_dim1=16 rows, tile_dim2=0
      g1[5] = 2048;                                  // tensor_dim0_stride[31:0]
      g1[6] = 0;                                     // stride[47:32], dim1_stride[15:0]
      g1[7] = 0;
      v4i gz = {0, 0, 0, 0};                         // 2D tensor: groups 2/3 unused
      TDM_LOAD(g0, g1, gz, gz);
    }
  };
  issue(0);
#else
  // Fallback: cooperative copy with plain loads, same double-buffer structure.
  auto issue = [&](int c) {
    const bf16_t* src = Abase + c * KC;
    bf16_t* dst = &sA[c & 1][0];
    for (int i = tid; i < 512; i += 256) {           // 512 x 16B = 16 rows x 512B
      const int row = i >> 5, seg = i & 31;
      *(v8bf*)(dst + row * LDB_B + seg * 8) = *(const v8bf*)(src + (size_t)row * 2048 + seg * 8);
    }
  };
  issue(0);
#endif

  v8f acc0 = {}; v8f acc1 = {};
  for (int c = 0; c < 2048 / KC; ++c) {
#if USE_TDM
    if (wave == 0) __builtin_amdgcn_s_wait_tensorcnt(0);   // TENSORcnt is per-wave (issuer)
#endif
    __syncthreads();                     // chunk c resident in sA[c&1] for all waves
    if (c + 1 < 2048 / KC) issue(c + 1); // prefetch next chunk into the other buffer
    const bf16_t* Ach = &sA[c & 1][0];
    #pragma unroll
    for (int kk = 0; kk < KC; kk += 32) {
      const int kb = c * KC + kk;
      v16bf a  = load_a_frag(Ach, LDB_B, lane, kk);
      v16bf b0 = load_b_frag(Wpb, D_MODEL, lane, kb, nbase);
      v16bf b1 = load_b_frag(Wpb, D_MODEL, lane, kb, nbase + 16);
      acc0 = __builtin_amdgcn_wmma_f32_16x16x32_bf16(false, a, false, b0, (short)0, acc0, false, false);
      acc1 = __builtin_amdgcn_wmma_f32_16x16x32_bf16(false, a, false, b1, (short)0, acc1, false, false);
    }
    __syncthreads();                     // all reads of sA[c&1] done before it is rewritten
  }

  const int col = nbase + (lane & 15);
  const int rb  = (lane >> 4) << 3;
  const float bi0 = bp[col];
  const float bi1 = bp[col + 16];
  #pragma unroll
  for (int r = 0; r < 8; ++r) {
    out[(size_t)(m0 + rb + r) * D_MODEL + col]      = acc0[r] + bi0;
    out[(size_t)(m0 + rb + r) * D_MODEL + col + 16] = acc1[r] + bi1;
  }
}

extern "C" void kernel_launch(void* const* d_in, const int* in_sizes, int n_in,
                              void* d_out, int out_size, void* d_ws, size_t ws_size,
                              hipStream_t stream) {
  const float* spec = (const float*)d_in[0];
  const float* W1   = (const float*)d_in[1];
  const float* b1   = (const float*)d_in[2];
  const float* W2   = (const float*)d_in[3];
  const float* b2   = (const float*)d_in[4];
  const float* Wq   = (const float*)d_in[5];
  const float* bq   = (const float*)d_in[6];
  const float* Wk   = (const float*)d_in[7];
  const float* bk   = (const float*)d_in[8];
  const float* Wv   = (const float*)d_in[9];
  const float* bv   = (const float*)d_in[10];
  const float* Wo   = (const float*)d_in[11];
  const float* bo   = (const float*)d_in[12];
  const float* Wp   = (const float*)d_in[13];
  const float* bp   = (const float*)d_in[14];

  const int B = 64, S = 3200;
  const int T  = B * S;          // 204800 tokens
  const int NW = T / PATCH;      // 25600 windows

  // Workspace: 5 x 256x256 bf16 weights, Wp bf16, then O [T,256] bf16 (~107 MB total)
  bf16_t* W2b  = (bf16_t*)d_ws;
  bf16_t* Wqb  = W2b + 65536;
  bf16_t* Wkb  = Wqb + 65536;
  bf16_t* Wvb  = Wkb + 65536;
  bf16_t* Wob  = Wvb + 65536;
  bf16_t* Wpb  = Wob + 65536;
  bf16_t* Obuf = Wpb + 2048 * 256;

  cvt_f32_bf16_kernel<<<(65536 + 255) / 256, 256, 0, stream>>>(W2, W2b, 65536);
  cvt_f32_bf16_kernel<<<(65536 + 255) / 256, 256, 0, stream>>>(Wq, Wqb, 65536);
  cvt_f32_bf16_kernel<<<(65536 + 255) / 256, 256, 0, stream>>>(Wk, Wkb, 65536);
  cvt_f32_bf16_kernel<<<(65536 + 255) / 256, 256, 0, stream>>>(Wv, Wvb, 65536);
  cvt_f32_bf16_kernel<<<(65536 + 255) / 256, 256, 0, stream>>>(Wo, Wob, 65536);
  cvt_f32_bf16_kernel<<<(524288 + 255) / 256, 256, 0, stream>>>(Wp, Wpb, 524288);

  fused_embed_attn_kernel<<<T / 16, 256, 0, stream>>>(
      spec, W1, b1, b2, bq, bk, bv, bo, W2b, Wqb, Wkb, Wvb, Wob, Obuf);

  patch_proj_kernel<<<NW / 16, 256, 0, stream>>>(Obuf, Wpb, bp, (float*)d_out);
}
